// EquivariantGNN_GAT_54211077210112
// MI455X (gfx1250) — matrix operand order repaired and checked by reading, not compiled, not run on hardware
//
#include <hip/hip_runtime.h>

// ---------------- problem constants ----------------
constexpr int N_NODES = 50000;
constexpr int N_EDGES = 1600000;
constexpr int E_TOT   = N_EDGES + N_NODES;   // with self loops
constexpr int FDIM    = 128;                 // 3 + 125 = hidden = 128
constexpr int ODIM    = 32;
constexpr int FEMB    = 125;
constexpr int NGRAPH  = 64;
constexpr float NEG_SLOPE = 0.2f;

constexpr int ROWS_PER_BLK = 80;             // 50000 = 625 * 80 (exact, no tail)
constexpr int MTILES       = ROWS_PER_BLK / 16;   // 5
constexpr int GEMM_BLOCKS  = N_NODES / ROWS_PER_BLK; // 625

typedef __attribute__((ext_vector_type(16))) _Float16 v16h;
typedef __attribute__((ext_vector_type(8)))  _Float16 v8h;
typedef __attribute__((ext_vector_type(8)))  float    v8f;

// ---------------- helpers ----------------
__device__ inline void atomic_add_f32(float* p, float v) {
    unsafeAtomicAdd(p, v);   // global_atomic_add_f32 (L2-resident here)
}

// float atomic max via monotonic int-bit trick (valid with -inf init)
__device__ inline void atomic_max_f32(float* addr, float v) {
    if (v >= 0.0f) atomicMax((int*)addr, __float_as_int(v));
    else           atomicMin((unsigned int*)addr, __float_as_uint(v));
}

__device__ inline void edge_sd(const int* __restrict__ ei, int e, int& s, int& d) {
    if (e < N_EDGES) { s = ei[e]; d = ei[N_EDGES + e]; }
    else             { s = d = e - N_EDGES; }           // self loop
}

// ---------------- weight prep: W[K][Ncol] f32 -> Wt[Ncol][K] f16 ----------------
__global__ void prep_w_kernel(const float* __restrict__ W, _Float16* __restrict__ Wt,
                              int K, int Ncol) {
    int i = blockIdx.x * blockDim.x + threadIdx.x;
    if (i >= K * Ncol) return;
    int k = i / Ncol, n = i - k * Ncol;
    Wt[n * K + k] = (_Float16)W[i];
}

// ---------------- x = concat(pos, emb[z]) as f16 [N,128] ----------------
__global__ void build_x_kernel(const float* __restrict__ pos, const int* __restrict__ z,
                               const float* __restrict__ emb, _Float16* __restrict__ x) {
    int i = blockIdx.x * blockDim.x + threadIdx.x;
    if (i >= N_NODES * FDIM) return;
    int node = i >> 7, f = i & 127;
    float v = (f < 3) ? pos[node * 3 + f] : emb[z[node] * FEMB + (f - 3)];
    x[i] = (_Float16)v;
}

// ---------------- WMMA GEMM: C[N,NCOLS] = A[N,128]f16 * Bt[NCOLS,128]^T ----------
// Block: NCOLS/16 waves, 80 rows (5 M-tiles). B fragments held in registers across
// all 5 M-tiles; 80x128 f16 A tile staged via async global->LDS (contiguous 20KB).
template<int NCOLS, bool BIAS>
__global__ void gemm_wmma_kernel(const _Float16* __restrict__ A,
                                 const _Float16* __restrict__ Bt,
                                 const float*    __restrict__ bias,
                                 float*          __restrict__ C) {
    __shared__ __attribute__((aligned(16))) _Float16 As[ROWS_PER_BLK * FDIM];
    const int tid = threadIdx.x;
    const int rowBase = blockIdx.x * ROWS_PER_BLK;

    // ---- async stage: A tile is one contiguous chunk starting at rowBase*256 bytes
    {
        const unsigned ldsBase = (unsigned)(uintptr_t)&As[0];      // low 32 bits = LDS offset
        const unsigned gBase   = (unsigned)rowBase * (FDIM * 2);   // byte offset into A
        constexpr int CHUNKS   = ROWS_PER_BLK * FDIM * 2 / 16;     // 16B per lane-op
        for (int i = tid; i < CHUNKS; i += blockDim.x) {
            unsigned ldsOff = ldsBase + (unsigned)i * 16u;
            unsigned vOff   = gBase   + (unsigned)i * 16u;
            asm volatile("global_load_async_to_lds_b128 %0, %1, %2"
                         :: "v"(ldsOff), "v"(vOff), "s"(A) : "memory");
        }
        asm volatile("s_wait_asynccnt 0x0" ::: "memory");
    }
    __syncthreads();

    const int wave   = tid >> 5;
    const int lane   = tid & 31;
    const int laneLo = lane & 15;
    const int laneHi = lane >> 4;
    const int colBase = wave * 16;
    const int col     = colBase + laneLo;

    // ---- B fragments (32x16 each), loaded once, reused across all 5 M-tiles
    v16h bfrag[FDIM / 32];
#pragma unroll
    for (int kk = 0; kk < FDIM / 32; ++kk) {
        const _Float16* bp = &Bt[col * FDIM + kk * 32 + laneHi * 8];
        v8h blo = *(const v8h*)bp;
        v8h bhi = *(const v8h*)(bp + 16);
#pragma unroll
        for (int i = 0; i < 8; ++i) { bfrag[kk][i] = blo[i]; bfrag[kk][8 + i] = bhi[i]; }
    }

    const float bv = BIAS ? bias[col] : 0.0f;

#pragma unroll
    for (int t = 0; t < MTILES; ++t) {
        v8f acc = {};
#pragma unroll
        for (int kk = 0; kk < FDIM / 32; ++kk) {
            // A 16x32 fragment: lanes 0-15 row=lane K={0..7,16..23}; 16-31 K={8..15,24..31}
            const _Float16* ap = &As[(t * 16 + laneLo) * FDIM + kk * 32 + laneHi * 8];
            v8h alo = *(const v8h*)ap;
            v8h ahi = *(const v8h*)(ap + 16);
            v16h a;
#pragma unroll
            for (int i = 0; i < 8; ++i) { a[i] = alo[i]; a[8 + i] = ahi[i]; }
            acc = __builtin_amdgcn_wmma_f32_16x16x32_f16(false, a, false, bfrag[kk],
                                                         (short)0, acc, false, false);
        }
        const int rowT = rowBase + t * 16 + laneHi * 8;   // D layout: M = r + 8*hi, N = laneLo
#pragma unroll
        for (int r = 0; r < 8; ++r)
            C[(rowT + r) * NCOLS + col] = acc[r] + bv;
    }
}

// ---------------- per-node attention dots: s_src = h.a_s, s_dst = h.a_d ----------
__global__ void node_dots_kernel(const float* __restrict__ h,
                                 const float* __restrict__ a_s, const float* __restrict__ a_d,
                                 float* __restrict__ s_src, float* __restrict__ s_dst) {
    int wid  = (blockIdx.x * blockDim.x + threadIdx.x) >> 5;
    int lane = threadIdx.x & 31;
    if (wid >= N_NODES) return;
    float4 hv = ((const float4*)(h + wid * FDIM))[lane];
    float4 as = ((const float4*)a_s)[lane];
    float4 ad = ((const float4*)a_d)[lane];
    float ps = hv.x * as.x + hv.y * as.y + hv.z * as.z + hv.w * as.w;
    float pd = hv.x * ad.x + hv.y * ad.y + hv.z * ad.z + hv.w * ad.w;
#pragma unroll
    for (int off = 16; off > 0; off >>= 1) {
        ps += __shfl_xor(ps, off, 32);
        pd += __shfl_xor(pd, off, 32);
    }
    if (lane == 0) { s_src[wid] = ps; s_dst[wid] = pd; }
}

__global__ void set_neginf_kernel(float* __restrict__ p, int n) {
    int i = blockIdx.x * blockDim.x + threadIdx.x;
    if (i < n) p[i] = __int_as_float(0xff800000);
}

// ---------------- edge passes ----------------
__global__ void edge_score_kernel(const int* __restrict__ ei,
                                  const float* __restrict__ s_src, const float* __restrict__ s_dst,
                                  float* __restrict__ score, float* __restrict__ m) {
    int e = blockIdx.x * blockDim.x + threadIdx.x;
    if (e >= E_TOT) return;
    int s, d; edge_sd(ei, e, s, d);
    float v = s_src[s] + s_dst[d];
    v = (v > 0.0f) ? v : NEG_SLOPE * v;               // leaky relu
    score[e] = v;
    atomic_max_f32(&m[d], v);
}

__global__ void edge_exp_kernel(const int* __restrict__ ei,
                                float* __restrict__ score, const float* __restrict__ m,
                                float* __restrict__ denom) {
    int e = blockIdx.x * blockDim.x + threadIdx.x;
    if (e >= E_TOT) return;
    int s, d; edge_sd(ei, e, s, d);
    float w = __expf(score[e] - m[d]);
    score[e] = w;                                     // in-place: score -> exp weight
    atomic_add_f32(&denom[d], w);
}

// wave per edge: 32 lanes x float4 = 128 features
__global__ void edge_agg_kernel(const int* __restrict__ ei,
                                const float* __restrict__ ew, const float* __restrict__ denom,
                                const float* __restrict__ h, float* __restrict__ out) {
    int wid  = (blockIdx.x * blockDim.x + threadIdx.x) >> 5;
    int lane = threadIdx.x & 31;
    if (wid >= E_TOT) return;
    int s, d; edge_sd(ei, wid, s, d);
    float alpha = ew[wid] / denom[d];
    float4 hv = ((const float4*)(h + s * FDIM))[lane];
    float* op = out + d * FDIM + lane * 4;
    atomic_add_f32(op + 0, alpha * hv.x);
    atomic_add_f32(op + 1, alpha * hv.y);
    atomic_add_f32(op + 2, alpha * hv.z);
    atomic_add_f32(op + 3, alpha * hv.w);
}

// ---------------- x_next = f16(elu(agg + bias)) ----------------
__global__ void post_elu_kernel(const float* __restrict__ agg, const float* __restrict__ b,
                                _Float16* __restrict__ xo) {
    int i = blockIdx.x * blockDim.x + threadIdx.x;
    if (i >= N_NODES * FDIM) return;
    float v = agg[i] + b[i & 127];
    v = (v > 0.0f) ? v : (__expf(v) - 1.0f);
    xo[i] = (_Float16)v;
}

// ---------------- graph mean pool ----------------
__global__ void pool_kernel(const float* __restrict__ y, const int* __restrict__ batch,
                            float* __restrict__ sums, float* __restrict__ cnt) {
    int wid  = (blockIdx.x * blockDim.x + threadIdx.x) >> 5;
    int lane = threadIdx.x & 31;
    if (wid >= N_NODES) return;
    int g = batch[wid];
    atomic_add_f32(&sums[g * ODIM + lane], y[wid * ODIM + lane]);   // ODIM == 32 lanes
    if (lane == 0) atomic_add_f32(&cnt[g], 1.0f);
}

__global__ void finalize_kernel(const float* __restrict__ sums, const float* __restrict__ cnt,
                                float* __restrict__ out) {
    int i = blockIdx.x * blockDim.x + threadIdx.x;
    if (i >= NGRAPH * ODIM) return;
    out[i] = sums[i] / fmaxf(cnt[i >> 5], 1.0f);
}

// ---------------- host ----------------
extern "C" void kernel_launch(void* const* d_in, const int* in_sizes, int n_in,
                              void* d_out, int out_size, void* d_ws, size_t ws_size,
                              hipStream_t stream) {
    const float* pos   = (const float*)d_in[0];
    const int*   z     = (const int*)  d_in[1];
    const int*   ei    = (const int*)  d_in[2];
    const int*   batch = (const int*)  d_in[3];
    const float* emb   = (const float*)d_in[4];
    const float* W1    = (const float*)d_in[5];
    const float* a1s   = (const float*)d_in[6];
    const float* a1d   = (const float*)d_in[7];
    const float* b1    = (const float*)d_in[8];
    const float* W2    = (const float*)d_in[9];
    const float* a2s   = (const float*)d_in[10];
    const float* a2d   = (const float*)d_in[11];
    const float* b2    = (const float*)d_in[12];
    const float* Wlin  = (const float*)d_in[13];
    const float* blin  = (const float*)d_in[14];

    char* ws = (char*)d_ws;
    size_t off = 0;
    auto alloc = [&](size_t bytes) -> void* {
        void* p = ws + off;
        off = (off + bytes + 255) & ~size_t(255);
        return p;
    };
    _Float16* x16   = (_Float16*)alloc((size_t)N_NODES * FDIM * 2);
    _Float16* wt1   = (_Float16*)alloc((size_t)FDIM * FDIM * 2);
    _Float16* wt2   = (_Float16*)alloc((size_t)FDIM * FDIM * 2);
    _Float16* wtl   = (_Float16*)alloc((size_t)FDIM * ODIM * 2);
    float*    h     = (float*)alloc((size_t)N_NODES * FDIM * 4);
    float*    agg   = (float*)alloc((size_t)N_NODES * FDIM * 4);
    float*    ssrc  = (float*)alloc((size_t)N_NODES * 4);
    float*    sdst  = (float*)alloc((size_t)N_NODES * 4);
    float*    m     = (float*)alloc((size_t)N_NODES * 4);
    float*    denom = (float*)alloc((size_t)N_NODES * 4);
    float*    score = (float*)alloc((size_t)E_TOT * 4);
    float*    y     = (float*)alloc((size_t)N_NODES * ODIM * 4);
    float*    sums  = (float*)alloc((size_t)(NGRAPH * ODIM + NGRAPH) * 4);
    float*    cnt   = sums + NGRAPH * ODIM;

    const int T256 = 256;
    auto blk = [](long long n, int t) { return (unsigned)((n + t - 1) / t); };

    // weights -> transposed f16
    prep_w_kernel<<<blk(FDIM * FDIM, T256), T256, 0, stream>>>(W1, wt1, FDIM, FDIM);
    prep_w_kernel<<<blk(FDIM * FDIM, T256), T256, 0, stream>>>(W2, wt2, FDIM, FDIM);
    prep_w_kernel<<<blk(FDIM * ODIM, T256), T256, 0, stream>>>(Wlin, wtl, FDIM, ODIM);

    // x = concat(pos, emb[z])
    build_x_kernel<<<blk((long long)N_NODES * FDIM, T256), T256, 0, stream>>>(pos, z, emb, x16);

    auto gat_layer = [&](const _Float16* wt, const float* a_s, const float* a_d,
                         const float* bias) {
        gemm_wmma_kernel<FDIM, false><<<GEMM_BLOCKS, (FDIM / 16) * 32, 0, stream>>>(
            x16, wt, nullptr, h);
        node_dots_kernel<<<blk((long long)N_NODES * 32, T256), T256, 0, stream>>>(
            h, a_s, a_d, ssrc, sdst);
        set_neginf_kernel<<<blk(N_NODES, T256), T256, 0, stream>>>(m, N_NODES);
        hipMemsetAsync(denom, 0, (size_t)N_NODES * 4, stream);
        hipMemsetAsync(agg,   0, (size_t)N_NODES * FDIM * 4, stream);
        edge_score_kernel<<<blk(E_TOT, T256), T256, 0, stream>>>(ei, ssrc, sdst, score, m);
        edge_exp_kernel<<<blk(E_TOT, T256), T256, 0, stream>>>(ei, score, m, denom);
        edge_agg_kernel<<<blk((long long)E_TOT * 32, T256), T256, 0, stream>>>(
            ei, score, denom, h, agg);
        post_elu_kernel<<<blk((long long)N_NODES * FDIM, T256), T256, 0, stream>>>(
            agg, bias, x16);
    };

    gat_layer(wt1, a1s, a1d, b1);
    gat_layer(wt2, a2s, a2d, b2);

    // final linear [N,32] with bias
    gemm_wmma_kernel<ODIM, true><<<GEMM_BLOCKS, (ODIM / 16) * 32, 0, stream>>>(
        x16, wtl, blin, y);

    // mean pool per graph
    hipMemsetAsync(sums, 0, (size_t)(NGRAPH * ODIM + NGRAPH) * 4, stream);
    pool_kernel<<<blk((long long)N_NODES * 32, T256), T256, 0, stream>>>(y, batch, sums, cnt);
    finalize_kernel<<<blk(NGRAPH * ODIM, T256), T256, 0, stream>>>(sums, cnt, (float*)d_out);
}
